// MPNNModel_14869176778848
// MI455X (gfx1250) — matrix operand order repaired and checked by reading, not compiled, not run on hardware
//
#include <hip/hip_runtime.h>

typedef __bf16 bf16;
typedef __attribute__((ext_vector_type(16))) __bf16 v16bf;
typedef __attribute__((ext_vector_type(8)))  __bf16 v8bf;
typedef __attribute__((ext_vector_type(8)))  float   v8f;

#define HD 128

// ---------------------------------------------------------------------------
// C[M,128] = relu(A[M,128](bf16) @ W[128,128](bf16, row-major)^T + bias)
// Block: 256 threads = 8 waves; each wave: TWO 16-row M tiles (32 rows), all
// 128 cols. W + bias staged in LDS. Per K-step: 8 B-fragments are batch-loaded
// from LDS (16 ds_load_b128 in one clause), then 16 WMMAs consume them, so the
// scheduler can overlap LDS latency with v_wmma_f32_16x16x32_bf16 issue.
// 64 WMMAs / 64 ds_load_b128 per wave.
// ---------------------------------------------------------------------------
__global__ __launch_bounds__(256)
void gemm_relu_bf16_wmma(const bf16* __restrict__ A, const bf16* __restrict__ W,
                         const float* __restrict__ bias, float* __restrict__ C, int M)
{
    __shared__ __align__(16) bf16 ldsW[HD * HD];
    __shared__ float ldsB[HD];

    const int tid = threadIdx.x;
    {   // cooperative stage of W (32KB) into LDS
        const v8bf* src = reinterpret_cast<const v8bf*>(W);
        v8bf* dst = reinterpret_cast<v8bf*>(ldsW);
        #pragma unroll
        for (int i = 0; i < (HD * HD / 8) / 256; ++i)
            dst[tid + i * 256] = src[tid + i * 256];
        if (tid < HD) ldsB[tid] = bias[tid];
    }
    __syncthreads();

    const int wave = tid >> 5;
    const int lane = tid & 31;
    const int hi   = lane >> 4;   // 0 for lanes 0-15, 1 for lanes 16-31
    const int r    = lane & 15;
    const int m0   = (blockIdx.x * 8 + wave) * 32;   // 2 tiles per wave
    if (m0 >= M) return;          // wave-uniform exit (no barriers after this)

    // ---- A fragments, 2 tiles x 4 K-steps: ISA 16-bit A 16x32 layout ----
    // lanes 0-15 : row r, K in {k0+0..7, k0+16..23}
    // lanes 16-31: row r, K in {k0+8..15, k0+24..31}
    v16bf afr[2][4];
    #pragma unroll
    for (int t = 0; t < 2; ++t) {
        int rowa = m0 + t * 16 + r;
        if (rowa > M - 1) rowa = M - 1;               // clamp tail (stores guarded)
        const bf16* arow = A + (long)rowa * HD + hi * 8;
        #pragma unroll
        for (int ks = 0; ks < 4; ++ks) {
            v8bf lo  = *reinterpret_cast<const v8bf*>(arow + ks * 32);
            v8bf hi8 = *reinterpret_cast<const v8bf*>(arow + ks * 32 + 16);
            afr[t][ks] = __builtin_shufflevector(lo, hi8,
                0,1,2,3,4,5,6,7,8,9,10,11,12,13,14,15);
        }
    }

    // ---- main loop: per K-step, batch-load 8 B frags, then 16 WMMAs ----
    // B layout: lane L holds column n=L%16, K in {k0+0..7, k0+16..23} (k0 += 8 for lanes>=16)
    v8f acc0[8] = {};
    v8f acc1[8] = {};
    #pragma unroll
    for (int ks = 0; ks < 4; ++ks) {
        v16bf bfr[8];
        #pragma unroll
        for (int nt = 0; nt < 8; ++nt) {
            const bf16* wrow = ldsW + (nt * 16 + r) * HD + hi * 8 + ks * 32;
            v8bf blo = *reinterpret_cast<const v8bf*>(wrow);
            v8bf bhi = *reinterpret_cast<const v8bf*>(wrow + 16);
            bfr[nt] = __builtin_shufflevector(blo, bhi,
                0,1,2,3,4,5,6,7,8,9,10,11,12,13,14,15);
        }
        #pragma unroll
        for (int nt = 0; nt < 8; ++nt)
            acc0[nt] = __builtin_amdgcn_wmma_f32_16x16x32_bf16(
                false, afr[0][ks], false, bfr[nt], (short)0, acc0[nt], false, false);
        #pragma unroll
        for (int nt = 0; nt < 8; ++nt)
            acc1[nt] = __builtin_amdgcn_wmma_f32_16x16x32_bf16(
                false, afr[1][ks], false, bfr[nt], (short)0, acc1[nt], false, false);
    }

    // ---- epilogue: bias + relu + store (D layout: vgpr rr -> row +rr+hi*8, col=lane%16) ----
    #pragma unroll
    for (int nt = 0; nt < 8; ++nt) {
        const int col = nt * 16 + r;
        const float bv = ldsB[col];
        #pragma unroll
        for (int rr = 0; rr < 8; ++rr) {
            int row0 = m0 + rr + hi * 8;
            if (row0 < M) {
                float v = acc0[nt][rr] + bv;
                C[(long)row0 * HD + col] = v > 0.f ? v : 0.f;
            }
            int row1 = m0 + 16 + rr + hi * 8;
            if (row1 < M) {
                float v = acc1[nt][rr] + bv;
                C[(long)row1 * HD + col] = v > 0.f ? v : 0.f;
            }
        }
    }
}

// ---------------------------------------------------------------------------
// helper kernels
// ---------------------------------------------------------------------------
__global__ void k_f32_to_bf16(const float* __restrict__ in, bf16* __restrict__ out, int n) {
    int i = blockIdx.x * blockDim.x + threadIdx.x;
    if (i < n) out[i] = (bf16)in[i];
}

// out[r][0:rlen]=a[r], out[r][rlen:2rlen]=b[r]  (f32 -> bf16)
__global__ void k_pack_pair(const float* __restrict__ a, const float* __restrict__ b,
                            bf16* __restrict__ out, int rows, int rlen) {
    int i = blockIdx.x * blockDim.x + threadIdx.x;
    int rl2 = rlen * 2;
    if (i >= rows * rl2) return;
    int r = i / rl2, c = i % rl2;
    float v = (c < rlen) ? a[r * rlen + c] : b[r * rlen + (c - rlen)];
    out[i] = (bf16)v;
}

__global__ void k_vec_add2(const float* __restrict__ a, const float* __restrict__ b,
                           float* __restrict__ out, int n) {
    int i = blockIdx.x * blockDim.x + threadIdx.x;
    if (i < n) out[i] = a[i] + b[i];
}

__global__ void k_zero(float* __restrict__ p, int n) {
    int i = blockIdx.x * blockDim.x + threadIdx.x;
    if (i < n) p[i] = 0.f;
}

// per-edge scatter-sum: agg[dst] += m[src]; 32 lanes x float4 = 128 floats/edge
__global__ void k_scatter_add(const float* __restrict__ m, const int* __restrict__ epair,
                              float* __restrict__ agg, int E_) {
    int t = blockIdx.x * blockDim.x + threadIdx.x;
    int e = t >> 5;
    if (e >= E_) return;
    int lane = t & 31;
    int s = epair[e];
    int d = epair[E_ + e];
    float4 v = *(reinterpret_cast<const float4*>(m + (long)s * HD) + lane);
    float* o = agg + (long)d * HD + lane * 4;
    atomicAdd(o + 0, v.x); atomicAdd(o + 1, v.y);
    atomicAdd(o + 2, v.z); atomicAdd(o + 3, v.w);
}

// segment-sum into only 2 destinations (t nodes): register-accumulate per wave,
// then a handful of block-level atomics (avoids 38M same-address atomics).
__global__ __launch_bounds__(256)
void k_seg2_reduce(const float* __restrict__ m, const int* __restrict__ epair,
                   float* __restrict__ agg_t, int E_) {
    const int lane = threadIdx.x & 31;
    const int w    = blockIdx.x * 8 + (threadIdx.x >> 5);
    const int nw   = gridDim.x * 8;
    float a0x = 0, a0y = 0, a0z = 0, a0w = 0;
    float a1x = 0, a1y = 0, a1z = 0, a1w = 0;
    for (int e = w; e < E_; e += nw) {
        int s = epair[e];
        int d = epair[E_ + e];
        float4 v = *(reinterpret_cast<const float4*>(m + (long)s * HD) + lane);
        if (d == 0) { a0x += v.x; a0y += v.y; a0z += v.z; a0w += v.w; }
        else        { a1x += v.x; a1y += v.y; a1z += v.z; a1w += v.w; }
    }
    float* o0 = agg_t + lane * 4;
    float* o1 = agg_t + HD + lane * 4;
    atomicAdd(o0 + 0, a0x); atomicAdd(o0 + 1, a0y); atomicAdd(o0 + 2, a0z); atomicAdd(o0 + 3, a0w);
    atomicAdd(o1 + 0, a1x); atomicAdd(o1 + 1, a1y); atomicAdd(o1 + 2, a1z); atomicAdd(o1 + 3, a1w);
}

// m_t = relu(h_t[2,128] @ W[128,128]^T + b)   (tiny: 256 threads)
__global__ void k_t_msg(const float* __restrict__ h_t, const float* __restrict__ W,
                        const float* __restrict__ b, float* __restrict__ m_t) {
    int i = threadIdx.x;           // 0..255
    int node = i >> 7, n = i & 127;
    const float* hv = h_t + node * HD;
    const float* wv = W + n * HD;
    float acc = b[n];
    #pragma unroll 8
    for (int k = 0; k < HD; ++k) acc += hv[k] * wv[k];
    m_t[i] = acc > 0.f ? acc : 0.f;
}

// h += agg (f32 master state in d_out) and refresh bf16 copy for next layer GEMMs
__global__ void k_add_cvt(float* __restrict__ h, const float* __restrict__ agg,
                          bf16* __restrict__ hb, int n) {
    int i = blockIdx.x * blockDim.x + threadIdx.x;
    if (i < n) { float v = h[i] + agg[i]; h[i] = v; hb[i] = (bf16)v; }
}

__global__ void k_add_t(float* __restrict__ h_t, const float* __restrict__ agg_t) {
    int i = threadIdx.x;           // 256 threads
    h_t[i] += agg_t[i];
}

// ---------------------------------------------------------------------------
extern "C" void kernel_launch(void* const* d_in, const int* in_sizes, int n_in,
                              void* d_out, int out_size, void* d_ws, size_t ws_size,
                              hipStream_t stream) {
    (void)n_in; (void)out_size; (void)ws_size;
    const int FD = 64;
    const int NC = in_sizes[0] / FD;
    const int NA = in_sizes[2] / FD;
    const int E_ = in_sizes[38] / 2;
    const int L  = in_sizes[14] / (HD * HD);

    const float* x_cl = (const float*)d_in[0];
    const float* x_cc = (const float*)d_in[1];
    const float* x_al = (const float*)d_in[2];
    const float* x_ac = (const float*)d_in[3];
    const float* pt_emb = (const float*)d_in[4];
    const float* ft_emb = (const float*)d_in[5];
    const float* cl_W = (const float*)d_in[6];  const float* cl_b = (const float*)d_in[7];
    const float* cc_W = (const float*)d_in[8];  const float* cc_b = (const float*)d_in[9];
    const float* al_W = (const float*)d_in[10]; const float* al_b = (const float*)d_in[11];
    const float* ac_W = (const float*)d_in[12]; const float* ac_b = (const float*)d_in[13];
    const float* Wf[8]; const float* Bf[8];
    for (int i = 0; i < 8; ++i) { Wf[i] = (const float*)d_in[14 + 2 * i]; Bf[i] = (const float*)d_in[15 + 2 * i]; }
    // edge index arrays [2,E] int32: src at [0..E), dst at [E..2E)
    const int* ed[8];
    for (int i = 0; i < 8; ++i) ed[i] = (const int*)d_in[38 + i];
    enum { CC = 0, CA, CT, AC, AA, AT, TC, TA };

    // node states live directly in d_out: [h_c | h_a | h_t] f32
    float* h_c = (float*)d_out;
    float* h_a = h_c + (long)NC * HD;
    float* h_t = h_a + (long)NA * HD;

    // workspace carve-out
    char* wp = (char*)d_ws;
    auto alloc = [&](size_t bytes) -> void* {
        void* p = (void*)wp; wp += (bytes + 255) & ~(size_t)255; return p;
    };
    bf16*  hbc   = (bf16*)alloc((size_t)NC * HD * 2);
    bf16*  hba   = (bf16*)alloc((size_t)NA * HD * 2);
    float* agg_c = (float*)alloc((size_t)NC * HD * 4);
    float* agg_a = (float*)alloc((size_t)NA * HD * 4);
    float* agg_t = (float*)alloc(2 * HD * 4);
    const int NMX = NC > NA ? NC : NA;
    float* mbuf  = (float*)alloc((size_t)NMX * HD * 4);
    float* m_t   = (float*)alloc(2 * HD * 4);
    bf16* Wb[8];
    for (int i = 0; i < 8; ++i) Wb[i] = (bf16*)alloc((size_t)L * HD * HD * 2);
    bf16*  WcatC = (bf16*)alloc(HD * HD * 2);
    bf16*  WcatA = (bf16*)alloc(HD * HD * 2);
    float* bsumC = (float*)alloc(HD * 4);
    float* bsumA = (float*)alloc(HD * 4);
    bf16*  xcat  = (bf16*)mbuf;  // alias: encoder input staging reuses mbuf

    dim3 blk(256);
    auto cdiv = [](long a, long b) { return (int)((a + b - 1) / b); };
    const int ROWS_PER_BLOCK = 256;   // 8 waves x 32 rows

    // layer weights f32 -> bf16 (once)
    for (int i = 0; i < 8; ++i)
        k_f32_to_bf16<<<cdiv((long)L * HD * HD, 256), blk, 0, stream>>>(Wf[i], Wb[i], L * HD * HD);

    // ---- encoders: concat two K=64 matmuls into one K=128 WMMA GEMM ----
    k_pack_pair<<<cdiv((long)NC * HD, 256), blk, 0, stream>>>(x_cc, x_cl, xcat, NC, FD);
    k_pack_pair<<<cdiv(HD * HD, 256), blk, 0, stream>>>(cc_W, cl_W, WcatC, HD, FD);
    k_vec_add2<<<1, HD, 0, stream>>>(cc_b, cl_b, bsumC, HD);
    gemm_relu_bf16_wmma<<<cdiv(NC, ROWS_PER_BLOCK), blk, 0, stream>>>(xcat, WcatC, bsumC, h_c, NC);
    k_f32_to_bf16<<<cdiv((long)NC * HD, 256), blk, 0, stream>>>(h_c, hbc, NC * HD);

    k_pack_pair<<<cdiv((long)NA * HD, 256), blk, 0, stream>>>(x_ac, x_al, xcat, NA, FD);
    k_pack_pair<<<cdiv(HD * HD, 256), blk, 0, stream>>>(ac_W, al_W, WcatA, HD, FD);
    k_vec_add2<<<1, HD, 0, stream>>>(ac_b, al_b, bsumA, HD);
    gemm_relu_bf16_wmma<<<cdiv(NA, ROWS_PER_BLOCK), blk, 0, stream>>>(xcat, WcatA, bsumA, h_a, NA);
    k_f32_to_bf16<<<cdiv((long)NA * HD, 256), blk, 0, stream>>>(h_a, hba, NA * HD);

    hipMemcpyAsync(h_t,      pt_emb, HD * sizeof(float), hipMemcpyDeviceToDevice, stream);
    hipMemcpyAsync(h_t + HD, ft_emb, HD * sizeof(float), hipMemcpyDeviceToDevice, stream);

    const int sgrid = cdiv((long)E_ * 32, 256);
    for (int l = 0; l < L; ++l) {
        k_zero<<<cdiv((long)NC * HD, 256), blk, 0, stream>>>(agg_c, NC * HD);
        k_zero<<<cdiv((long)NA * HD, 256), blk, 0, stream>>>(agg_a, NA * HD);
        k_zero<<<1, 256, 0, stream>>>(agg_t, 2 * HD);

        const long wlo = (long)l * HD * HD; const int blo = l * HD;
        // messages into c
        gemm_relu_bf16_wmma<<<cdiv(NC, ROWS_PER_BLOCK), blk, 0, stream>>>(hbc, Wb[CC] + wlo, Bf[CC] + blo, mbuf, NC);
        k_scatter_add<<<sgrid, blk, 0, stream>>>(mbuf, ed[CC], agg_c, E_);
        gemm_relu_bf16_wmma<<<cdiv(NA, ROWS_PER_BLOCK), blk, 0, stream>>>(hba, Wb[AC] + wlo, Bf[AC] + blo, mbuf, NA);
        k_scatter_add<<<sgrid, blk, 0, stream>>>(mbuf, ed[AC], agg_c, E_);
        k_t_msg<<<1, 256, 0, stream>>>(h_t, Wf[TC] + wlo, Bf[TC] + blo, m_t);
        k_scatter_add<<<sgrid, blk, 0, stream>>>(m_t, ed[TC], agg_c, E_);
        // messages into a
        gemm_relu_bf16_wmma<<<cdiv(NC, ROWS_PER_BLOCK), blk, 0, stream>>>(hbc, Wb[CA] + wlo, Bf[CA] + blo, mbuf, NC);
        k_scatter_add<<<sgrid, blk, 0, stream>>>(mbuf, ed[CA], agg_a, E_);
        gemm_relu_bf16_wmma<<<cdiv(NA, ROWS_PER_BLOCK), blk, 0, stream>>>(hba, Wb[AA] + wlo, Bf[AA] + blo, mbuf, NA);
        k_scatter_add<<<sgrid, blk, 0, stream>>>(mbuf, ed[AA], agg_a, E_);
        k_t_msg<<<1, 256, 0, stream>>>(h_t, Wf[TA] + wlo, Bf[TA] + blo, m_t);
        k_scatter_add<<<sgrid, blk, 0, stream>>>(m_t, ed[TA], agg_a, E_);
        // messages into t (2 destinations -> hierarchical reduction)
        gemm_relu_bf16_wmma<<<cdiv(NC, ROWS_PER_BLOCK), blk, 0, stream>>>(hbc, Wb[CT] + wlo, Bf[CT] + blo, mbuf, NC);
        k_seg2_reduce<<<64, blk, 0, stream>>>(mbuf, ed[CT], agg_t, E_);
        gemm_relu_bf16_wmma<<<cdiv(NA, ROWS_PER_BLOCK), blk, 0, stream>>>(hba, Wb[AT] + wlo, Bf[AT] + blo, mbuf, NA);
        k_seg2_reduce<<<64, blk, 0, stream>>>(mbuf, ed[AT], agg_t, E_);
        // residual update + refresh bf16 activations
        k_add_cvt<<<cdiv((long)NC * HD, 256), blk, 0, stream>>>(h_c, agg_c, hbc, NC * HD);
        k_add_cvt<<<cdiv((long)NA * HD, 256), blk, 0, stream>>>(h_a, agg_a, hba, NA * HD);
        k_add_t<<<1, 256, 0, stream>>>(h_t, agg_t);
    }
}
